// Bevformernet_60919816126556
// MI455X (gfx1250) — compile-verified, matrix operand order checked by reading, not run on hardware
//
#include <hip/hip_runtime.h>
#include <hip/hip_bf16.h>
#include <stdint.h>
#include <math.h>

// ---------------------------------------------------------------------------
// BEVFormer spatial cross attention, MI455X (gfx1250, wave32, WMMA + TDM).
// Pipeline: cast -> WMMA GEMMs (value/off/attn proj, tiles staged into LDS by
// the Tensor Data Mover) -> softmax -> bilinear sampling with camera masking
// -> WMMA GEMM output proj + residual.
// Offsets/attention depend only on q (broadcast over the 6 cameras in the
// reference), so they are computed once, not per camera.
// ---------------------------------------------------------------------------

typedef _Float16     v16h __attribute__((ext_vector_type(16)));
typedef float        v8f  __attribute__((ext_vector_type(8)));
typedef unsigned int uv4  __attribute__((ext_vector_type(4)));
typedef int          iv8  __attribute__((ext_vector_type(8)));
typedef int          iv4  __attribute__((ext_vector_type(4)));

#define C_DIM 128
#define HEADS 4
#define PTS   8
// LDS row pitch in halves: 128 data + 8 pad (16B) => 272B pitch.
// Fragment rows r then map to banks (r*68) % 64 = 4r -> conflict-free b128.
#define LDS_PITCH 136

// ---------------- elementwise conversion kernels ----------------

__global__ void add_cast_f16_kernel(const float* __restrict__ a,
                                    const float* __restrict__ b,
                                    _Float16* __restrict__ o, int n) {
  int i = blockIdx.x * blockDim.x + threadIdx.x;
  if (i < n) o[i] = (_Float16)(a[i] + b[i]);
}

__global__ void cast_f16_kernel(const float* __restrict__ a,
                                _Float16* __restrict__ o, int n) {
  int i = blockIdx.x * blockDim.x + threadIdx.x;
  if (i < n) o[i] = (_Float16)a[i];
}

// W: 128 x Ncols row-major (fp32)  ->  Wt: Ncols x 128 row-major (fp16)
__global__ void transpose_cast_f16_kernel(const float* __restrict__ W,
                                          _Float16* __restrict__ Wt, int Ncols) {
  int i = blockIdx.x * blockDim.x + threadIdx.x;
  int total = C_DIM * Ncols;
  if (i >= total) return;
  int nn = i / C_DIM;
  int k  = i % C_DIM;
  Wt[nn * C_DIM + k] = (_Float16)W[k * Ncols + nn];
}

// ---------------- TDM: stage a 2D f16 panel into LDS with row padding -------

// D# per CDNA5 ISA ch.8: group0 = {flags, lds_addr, global_addr, type=2},
// group1 = {mask/data_size/pad, tensor_dim0/1, tile_dim0/1, dim0_stride}.
// pad_interval=5 (insert pad every 64 DWORDs = one 128-half row),
// pad_amount=3 (4 DWORDs = 16B) -> LDS pitch 272B = LDS_PITCH halves.
__device__ __forceinline__ void tdm_load_panel_f16(const _Float16* g, uint32_t lds_off,
                                                   int tile_rows, int tensor_rows,
                                                   int cols /*=128*/) {
  uint64_t ga = (uint64_t)(uintptr_t)g;
  uv4 g0;
  g0[0] = 1u;                                           // count=1, load descriptor
  g0[1] = lds_off;                                      // lds_addr (bytes)
  g0[2] = (uint32_t)ga;                                 // global_addr[31:0]
  g0[3] = (uint32_t)((ga >> 32) & 0x01FFFFFFu)          // global_addr[56:32]
        | (2u << 30);                                   // type = 2 ("image")
  iv8 g1;
  g1[0] = (int)((1u << 16)                              // data_size = 1 (2 bytes)
              | (1u << 20)                              // pad_enable
              | (5u << 22)                              // pad_interval: 64 DWORDs
              | (3u << 25));                            // pad_amount: 4 DWORDs
  g1[1] = (int)((unsigned)(cols & 0xFFFF) << 16);       // tensor_dim0[15:0] @ 63:48
  g1[2] = (int)(((unsigned)cols >> 16)                  // tensor_dim0[31:16]
              | ((unsigned)(tensor_rows & 0xFFFF) << 16)); // tensor_dim1[15:0]
  g1[3] = (int)((((unsigned)tensor_rows >> 16) & 0xFFFFu)  // tensor_dim1[31:16]
              | ((unsigned)(cols & 0xFFFF) << 16));     // tile_dim0
  g1[4] = (int)(tile_rows & 0xFFFF);                    // tile_dim1 (tile_dim2=0)
  g1[5] = cols;                                         // tensor_dim0_stride[31:0]
  g1[6] = 0;
  g1[7] = 0;
  iv4 z4 = {0, 0, 0, 0};
  iv8 z8 = {0, 0, 0, 0, 0, 0, 0, 0};
  // 6-arg form on this toolchain: (g0, g1, g2, g3, g4, cpol)
  __builtin_amdgcn_tensor_load_to_lds(g0, g1, z4, z4, z8, 0);
}

// ---------------- WMMA GEMM: D = A(f16,Mx128) * Bt(f16,Nx128)^T + bias (+res)

// Block = 128 threads = 4 waves = 64 output rows; TDM stages the A tile and
// the whole Bt panel into padded LDS; each wave keeps its 4 A fragments in
// VGPRs and sweeps all N-tiles (A traffic amortized across Ncols/16 tiles).
// B fragments for a tile are batch-loaded (8x ds_load_b128, one wait), then
// the 4 WMMAs chain back-to-back.
template <bool HAS_RESID>
__global__ __launch_bounds__(128)
void gemm_wmma_f16_kernel(const _Float16* __restrict__ A,   // M x 128
                          const _Float16* __restrict__ Bt,  // Ncols x 128
                          const float* __restrict__ bias,   // Ncols
                          const float* __restrict__ resid,  // M x Ncols (if HAS_RESID)
                          float* __restrict__ D,
                          int M, int Ncols) {
  __shared__ _Float16 ldsB[128 * LDS_PITCH];  // 34 KB
  __shared__ _Float16 ldsA[64 * LDS_PITCH];   // 17 KB

  const int K = C_DIM;
  int wave = threadIdx.x >> 5;
  int lane = threadIdx.x & 31;
  int mblock = blockIdx.x * 64;

  if (wave == 0) {
    int arows = M - mblock;
    if (arows > 64) arows = 64;
    // tensor_rows clamps OOB: TDM zero-fills rows past the tensor extent.
    tdm_load_panel_f16(Bt, (uint32_t)(uintptr_t)ldsB, Ncols, Ncols, K);
    tdm_load_panel_f16(A + (size_t)mblock * K, (uint32_t)(uintptr_t)ldsA, 64, arows, K);
    __builtin_amdgcn_s_wait_tensorcnt(0);
  }
  __syncthreads();   // barrier before any early-out (all waves participate)

  int m0 = mblock + wave * 16;
  if (m0 >= M) return;

  int half = lane >> 4;   // 0: K 0..7/16..23 ; 1: K 8..15/24..31 (ISA A layout)
  int r    = lane & 15;   // row (A) / col (B) within 16x16 tile

  // A fragments for this wave's 16 rows, all 4 K-steps, from padded LDS.
  v16h afrag[4];
  {
    const _Float16* arow = ldsA + (wave * 16 + r) * LDS_PITCH + 8 * half;
#pragma unroll
    for (int ks = 0; ks < 4; ++ks) {
      *(uint4*)(&afrag[ks])     = *(const uint4*)(arow + ks * 32);
      *((uint4*)(&afrag[ks])+1) = *(const uint4*)(arow + ks * 32 + 16);
    }
  }

  const _Float16* brow0 = ldsB + r * LDS_PITCH + 8 * half;
  for (int n0 = 0; n0 < Ncols; n0 += 16) {
    const _Float16* brow = brow0 + n0 * LDS_PITCH;
    // Batch-load all 4 B fragments, then chain the 4 WMMAs.
    v16h bfrag[4];
#pragma unroll
    for (int ks = 0; ks < 4; ++ks) {
      *(uint4*)(&bfrag[ks])     = *(const uint4*)(brow + ks * 32);
      *((uint4*)(&bfrag[ks])+1) = *(const uint4*)(brow + ks * 32 + 16);
    }
    v8f acc = {};
#pragma unroll
    for (int ks = 0; ks < 4; ++ks) {
      acc = __builtin_amdgcn_wmma_f32_16x16x32_f16(
          /*neg_a=*/false, afrag[ks], /*neg_b=*/false, bfrag[ks],
          /*c_mod=*/(short)0, acc, /*reuse_a=*/false, /*reuse_b=*/false);
    }
    // C/D layout: VGPR v <-> row v + 8*(lane/16), col = lane%16.
    int col = n0 + r;
    int rbase = m0 + half * 8;
    float bvv = bias[col];
#pragma unroll
    for (int v = 0; v < 8; ++v) {
      int row = rbase + v;
      float val = acc[v] + bvv;
      if (HAS_RESID) val += resid[(size_t)row * Ncols + col];
      D[(size_t)row * Ncols + col] = val;
    }
  }
}

// ---------------- softmax over P=8 per (n, h) ----------------

__global__ void softmax8_kernel(const float* __restrict__ logits,  // N x 32
                                float* __restrict__ w, int NH) {
  int i = blockIdx.x * blockDim.x + threadIdx.x;   // over N*HEADS
  if (i >= NH) return;
  int n = i / HEADS, h = i % HEADS;
  const float* L = logits + (size_t)n * (HEADS * PTS) + h * PTS;
  float* W = w + (size_t)n * (HEADS * PTS) + h * PTS;
  float mx = L[0];
#pragma unroll
  for (int p = 1; p < PTS; ++p) mx = fmaxf(mx, L[p]);
  float s = 0.f, e[PTS];
#pragma unroll
  for (int p = 0; p < PTS; ++p) { e[p] = __expf(L[p] - mx); s += e[p]; }
  float inv = 1.0f / s;
#pragma unroll
  for (int p = 0; p < PTS; ++p) W[p] = e[p] * inv;
}

// ---------------- bilinear sampling + masked camera average ----------------

// One block (128 threads) per query n. thread = h*32 + cc.
__global__ __launch_bounds__(128)
void sample_kernel(const float* __restrict__ vproj,   // S x Mv x 128
                   const float* __restrict__ off,     // N x 64  ((h,p),xy)
                   const float* __restrict__ attnw,   // N x 32  (h,p)
                   const float* __restrict__ ref,     // S x N x 8 x 2
                   const unsigned char* __restrict__ mask,  // S x N x 8 (bool)
                   float* __restrict__ slots,         // N x 128
                   int N, int S,
                   const int* __restrict__ pHf, const int* __restrict__ pWf) {
  const int Hf = *pHf;
  const int Wf = *pWf;
  const int Mv = Hf * Wf;
  const float invWf = 1.0f / (float)Wf;
  const float invHf = 1.0f / (float)Hf;

  int n = blockIdx.x;
  int t = threadIdx.x;
  int h = t >> 5, cc = t & 31;

  float acc = 0.0f;
  float cnt = 0.0f;

  for (int s = 0; s < S; ++s) {
    const unsigned char* mk = mask + ((size_t)s * N + n) * 8;
    int any = 0;
#pragma unroll
    for (int d = 0; d < 8; ++d) any |= mk[d];
    if (!any) continue;
    cnt += 1.0f;

    const float* vbase = vproj + (size_t)s * Mv * C_DIM + h * 32 + cc;
    const float* rf = ref + ((size_t)s * N + n) * 16;   // 8 anchors x (x,y)
    float a = 0.0f;
#pragma unroll
    for (int p = 0; p < PTS; ++p) {
      // P==D: point p uses z-anchor d=p (P/D == 1 in reference reshape)
      float lx = rf[p * 2 + 0] + off[(size_t)n * 64 + (h * PTS + p) * 2 + 0] * invWf;
      float ly = rf[p * 2 + 1] + off[(size_t)n * 64 + (h * PTS + p) * 2 + 1] * invHf;
      float x = lx * (float)Wf - 0.5f;
      float y = ly * (float)Hf - 0.5f;
      float x0f = floorf(x), y0f = floorf(y);
      float fx = x - x0f, fy = y - y0f;
      int x0 = (int)x0f, y0 = (int)y0f;

      auto samp = [&](int yy, int xx) -> float {
        if (xx < 0 || xx >= Wf || yy < 0 || yy >= Hf) return 0.0f;
        return vbase[(size_t)(yy * Wf + xx) * C_DIM];
      };
      float g = samp(y0,     x0)     * (1.0f - fx) * (1.0f - fy)
              + samp(y0,     x0 + 1) * fx          * (1.0f - fy)
              + samp(y0 + 1, x0)     * (1.0f - fx) * fy
              + samp(y0 + 1, x0 + 1) * fx          * fy;
      a += attnw[(size_t)n * 32 + h * PTS + p] * g;
    }
    acc += a;
  }
  if (cnt < 1.0f) cnt = 1.0f;
  slots[(size_t)n * C_DIM + t] = acc / cnt;
}

// ---------------------------------------------------------------------------

static inline size_t align256(size_t x) { return (x + 255) & ~(size_t)255; }

extern "C" void kernel_launch(void* const* d_in, const int* in_sizes, int n_in,
                              void* d_out, int out_size, void* d_ws, size_t ws_size,
                              hipStream_t stream) {
  const float* query  = (const float*)d_in[0];
  const float* qpos   = (const float*)d_in[1];
  const float* value  = (const float*)d_in[2];
  const float* ref    = (const float*)d_in[3];
  const unsigned char* mask = (const unsigned char*)d_in[4];  // jax bool = 1B
  const float* Wv     = (const float*)d_in[5];
  const float* bv     = (const float*)d_in[6];
  const float* Wo     = (const float*)d_in[7];
  const float* bo     = (const float*)d_in[8];
  const float* Wa     = (const float*)d_in[9];
  const float* ba     = (const float*)d_in[10];
  const float* Wout   = (const float*)d_in[11];
  const float* bout   = (const float*)d_in[12];
  const int*   pHf    = (const int*)d_in[13];
  const int*   pWf    = (const int*)d_in[14];

  const int C  = C_DIM;
  const int N  = in_sizes[0] / C;          // 16384 queries
  const int S  = in_sizes[4] / (N * 8);    // 6 cameras (bev_mask = S*N*8)
  const int SM = in_sizes[2] / C;          // S * Hf*Wf rows of value

  // ---- workspace carve-up ----
  char* ws = (char*)d_ws;
  size_t off_b = 0;
  auto carve = [&](size_t bytes) { char* p = ws + off_b; off_b += align256(bytes); return p; };
  _Float16* qh     = (_Float16*)carve((size_t)N * C * 2);
  _Float16* valh   = (_Float16*)carve((size_t)SM * C * 2);
  _Float16* Wvt    = (_Float16*)carve((size_t)C * C * 2);
  _Float16* Wot    = (_Float16*)carve((size_t)64 * C * 2);
  _Float16* Wat    = (_Float16*)carve((size_t)32 * C * 2);
  _Float16* Woutt  = (_Float16*)carve((size_t)C * C * 2);
  float*    vproj  = (float*)carve((size_t)SM * C * 4);
  float*    offb   = (float*)carve((size_t)N * 64 * 4);
  float*    attnl  = (float*)carve((size_t)N * 32 * 4);
  float*    attnw  = (float*)carve((size_t)N * 32 * 4);
  float*    slots  = (float*)carve((size_t)N * C * 4);
  _Float16* slotsh = (_Float16*)carve((size_t)N * C * 2);
  (void)ws_size;

  const int TB = 256;

  // 1) q = query + query_pos (f16); value cast (f16)
  add_cast_f16_kernel<<<(N * C + TB - 1) / TB, TB, 0, stream>>>(query, qpos, qh, N * C);
  cast_f16_kernel<<<(SM * C + TB - 1) / TB, TB, 0, stream>>>(value, valh, SM * C);

  // 2) weight transposes (B fragment wants N x K row-major)
  transpose_cast_f16_kernel<<<(C * C  + TB - 1) / TB, TB, 0, stream>>>(Wv,   Wvt,   C);
  transpose_cast_f16_kernel<<<(C * 64 + TB - 1) / TB, TB, 0, stream>>>(Wo,   Wot,   64);
  transpose_cast_f16_kernel<<<(C * 32 + TB - 1) / TB, TB, 0, stream>>>(Wa,   Wat,   32);
  transpose_cast_f16_kernel<<<(C * C  + TB - 1) / TB, TB, 0, stream>>>(Wout, Woutt, C);

  // 3) WMMA GEMMs (TDM-staged LDS tiles)
  {  // value projection: (SM x 128) @ Wv + bv
    gemm_wmma_f16_kernel<false><<<(SM + 63) / 64, 128, 0, stream>>>(valh, Wvt, bv, nullptr,
                                                                    vproj, SM, C);
  }
  {  // sampling offsets: (N x 128) @ Wo + bo   (computed ONCE, not per camera)
    gemm_wmma_f16_kernel<false><<<(N + 63) / 64, 128, 0, stream>>>(qh, Wot, bo, nullptr,
                                                                   offb, N, 64);
  }
  {  // attention logits: (N x 128) @ Wa + ba
    gemm_wmma_f16_kernel<false><<<(N + 63) / 64, 128, 0, stream>>>(qh, Wat, ba, nullptr,
                                                                   attnl, N, 32);
  }

  // 4) softmax over P=8 per (n, h)
  softmax8_kernel<<<(N * HEADS + TB - 1) / TB, TB, 0, stream>>>(attnl, attnw, N * HEADS);

  // 5) bilinear sampling + masked camera average -> slots (N x 128)
  sample_kernel<<<N, 128, 0, stream>>>(vproj, offb, attnw, ref, mask, slots, N, S, pHf, pWf);

  // 6) output projection + residual: out = slots @ Wout + bout + query
  cast_f16_kernel<<<(N * C + TB - 1) / TB, TB, 0, stream>>>(slots, slotsh, N * C);
  gemm_wmma_f16_kernel<true><<<(N + 63) / 64, 128, 0, stream>>>(slotsh, Woutt, bout, query,
                                                                (float*)d_out, N, C);
}
